// InjectionModule_82360292868293
// MI455X (gfx1250) — matrix operand order repaired and checked by reading, not compiled, not run on hardware
//
#include <hip/hip_runtime.h>
#include <hip/hip_bf16.h>
#include <math.h>

#define D_MODEL 2048
#define D_GNN   512
#define MT      32     // rows per workgroup in the big kernel
#define NB      4      // 16-wide n-tiles per chunk per wave
#define NTHREADS 256

typedef __bf16 bf16_t;
typedef __attribute__((ext_vector_type(16))) __bf16        v16bf;
typedef __attribute__((ext_vector_type(8)))  float         v8f;
typedef __attribute__((ext_vector_type(4)))  unsigned int  v4u;

struct V32B { v4u lo; v4u hi; };

__device__ __forceinline__ v16bf pack2(v4u lo, v4u hi) {
  V32B t; t.lo = lo; t.hi = hi;
  return __builtin_bit_cast(v16bf, t);
}

__device__ __forceinline__ v8f wmma_bf16(v16bf a, v16bf b, v8f c) {
  // D = A(16x32 bf16) * B(32x16 bf16) + C(16x16 f32)
  return __builtin_amdgcn_wmma_f32_16x16x32_bf16(false, a, false, b,
                                                 (short)0, c, false, false);
}

// ---------------------------------------------------------------------------
// Kernel 1: hc = LayerNorm(h_change @ w_proj^T + b_proj) * g1 + b1   (4 x 2048)
// ---------------------------------------------------------------------------
__global__ __launch_bounds__(NTHREADS)
void proj_ln_kernel(const float* __restrict__ h_change,
                    const float* __restrict__ w_proj,
                    const float* __restrict__ b_proj,
                    const float* __restrict__ g1,
                    const float* __restrict__ b1,
                    float* __restrict__ hc)
{
  const int b = blockIdx.x;
  __shared__ float sX[D_GNN];
  __shared__ float sY[D_MODEL];
  __shared__ float sRed[2];
  for (int k = threadIdx.x; k < D_GNN; k += NTHREADS) sX[k] = h_change[b * D_GNN + k];
  if (threadIdx.x < 2) sRed[threadIdx.x] = 0.f;
  __syncthreads();
  float p1 = 0.f, p2 = 0.f;
  for (int j = 0; j < D_MODEL / NTHREADS; ++j) {
    const int d = threadIdx.x + j * NTHREADS;
    const float* wr = w_proj + (size_t)d * D_GNN;
    float acc = b_proj[d];
    for (int k = 0; k < D_GNN; ++k) acc = fmaf(sX[k], wr[k], acc);
    sY[d] = acc; p1 += acc; p2 += acc * acc;
  }
  atomicAdd(&sRed[0], p1);
  atomicAdd(&sRed[1], p2);
  __syncthreads();
  const float mu = sRed[0] * (1.f / D_MODEL);
  const float rs = rsqrtf(sRed[1] * (1.f / D_MODEL) - mu * mu + 1e-5f);
  for (int j = 0; j < D_MODEL / NTHREADS; ++j) {
    const int d = threadIdx.x + j * NTHREADS;
    hc[b * D_MODEL + d] = (sY[d] - mu) * rs * g1[d] + b1[d];
  }
}

// ---------------------------------------------------------------------------
// Kernel 2/3: out[b,d] = dot(in[b,:], W[d,:]) + bias[d]    (4 x 2048, K=2048)
// ---------------------------------------------------------------------------
__global__ __launch_bounds__(NTHREADS)
void small_gemm_kernel(const float* __restrict__ in,
                       const float* __restrict__ W,
                       const float* __restrict__ bias,
                       float* __restrict__ out)
{
  const int b = blockIdx.x;
  __shared__ float sX[D_MODEL];
  for (int k = threadIdx.x; k < D_MODEL; k += NTHREADS) sX[k] = in[b * D_MODEL + k];
  __syncthreads();
  for (int j = 0; j < D_MODEL / NTHREADS; ++j) {
    const int d = threadIdx.x + j * NTHREADS;
    const float* wr = W + (size_t)d * D_MODEL;
    float acc = bias[d];
    for (int k = 0; k < D_MODEL; ++k) acc = fmaf(sX[k], wr[k], acc);
    out[b * D_MODEL + d] = acc;
  }
}

// ---------------------------------------------------------------------------
// Kernel 4: split w_gate (2048 x 4096 f32) into wl/wa bf16 (2048 x 2048 each)
// ---------------------------------------------------------------------------
__global__ __launch_bounds__(NTHREADS)
void cvt_wgate_kernel(const float* __restrict__ w_gate,
                      bf16_t* __restrict__ wl,
                      bf16_t* __restrict__ wa)
{
  const size_t total = (size_t)D_MODEL * D_MODEL;
  const size_t stride = (size_t)gridDim.x * NTHREADS;
  for (size_t i = (size_t)blockIdx.x * NTHREADS + threadIdx.x; i < total; i += stride) {
    const size_t n = i >> 11;          // row (output feature)
    const size_t k = i & (D_MODEL - 1);
    wl[i] = (bf16_t)w_gate[n * (2 * D_MODEL) + k];
    wa[i] = (bf16_t)w_gate[n * (2 * D_MODEL) + D_MODEL + k];
  }
}

// ---------------------------------------------------------------------------
// Kernel 5: fused attn_out LN + dual bf16 WMMA GEMM gate + blend + final LN
// One workgroup = 32 rows x full 2048 cols, K = 2048, two GEMMs fused into
// one accumulator (gate logit). 8 waves: 2 m-tiles x 4 n-columns.
// ---------------------------------------------------------------------------
__global__ __launch_bounds__(NTHREADS)
void fused_gate_ln_kernel(const float* __restrict__ h_llm,
                          const float* __restrict__ attn,
                          const bf16_t* __restrict__ wl,
                          const bf16_t* __restrict__ wa,
                          const float* __restrict__ g2v,
                          const float* __restrict__ b2v,
                          const float* __restrict__ bgate,
                          const float* __restrict__ g3v,
                          const float* __restrict__ b3v,
                          float* __restrict__ out)
{
  __shared__ bf16_t sAh[MT * D_MODEL];   // 128 KB: bf16 h_llm rows
  __shared__ bf16_t sAo[MT * D_MODEL];   // 128 KB: bf16 attn_out rows
  __shared__ float sMu[MT], sRs[MT], sS1[MT], sS2[MT], sMu3[MT], sRs3[MT];

  const int tid   = threadIdx.x;
  const int lane  = tid & 31;
  const int wave  = tid >> 5;
  const int lhalf = lane >> 4;
  const int lmod  = lane & 15;
  const int rowBase = blockIdx.x * MT;
  const int batch   = rowBase >> 12;       // T = 4096 rows per batch
  const float* attnb = attn + batch * D_MODEL;

  if (tid < MT) { sS1[tid] = 0.f; sS2[tid] = 0.f; }

  // ---- pass 1: per-row mean / rstd of (h_llm + attn) ----
  for (int i = 0; i < MT / 8; ++i) {
    const int rl = wave * (MT / 8) + i;
    const float* hrow = h_llm + (size_t)(rowBase + rl) * D_MODEL;
    float s = 0.f, s2 = 0.f;
    for (int c = lane; c < D_MODEL; c += 32) {
      const float x = hrow[c] + attnb[c];
      s += x; s2 += x * x;
    }
    for (int m = 16; m >= 1; m >>= 1) {
      s  += __shfl_xor(s,  m, 32);
      s2 += __shfl_xor(s2, m, 32);
    }
    if (lane == 0) {
      const float mu = s * (1.f / D_MODEL);
      sMu[rl] = mu;
      sRs[rl] = rsqrtf(s2 * (1.f / D_MODEL) - mu * mu + 1e-5f);
    }
  }
  __syncthreads();

  // ---- pass 2: stage bf16 A matrices (h and attn_out) in LDS ----
  for (int i = 0; i < MT / 8; ++i) {
    const int rl = wave * (MT / 8) + i;
    const float mu = sMu[rl], rs = sRs[rl];
    const float* hrow = h_llm + (size_t)(rowBase + rl) * D_MODEL;
    for (int c = lane; c < D_MODEL; c += 32) {
      const float h  = hrow[c];
      const float ao = (h + attnb[c] - mu) * rs * g2v[c] + b2v[c];
      sAh[rl * D_MODEL + c] = (bf16_t)h;
      sAo[rl * D_MODEL + c] = (bf16_t)ao;
    }
  }
  __syncthreads();

  // ---- WMMA GEMM: gate logits = h @ wl^T + attn_out @ wa^T ----
  const int mt  = wave & 1;      // which 16-row tile
  const int wc  = wave >> 1;     // which 512-col band
  const int arow = mt * 16 + lmod;
  const int akb  = lhalf * 16;   // A lane-half K offset (contiguous, matches B)
  const int bkb  = lhalf * 16;   // B lane-half K offset (contiguous, matches A)

  float s1acc[8], s2acc[8];
  #pragma unroll
  for (int v = 0; v < 8; ++v) { s1acc[v] = 0.f; s2acc[v] = 0.f; }

  const v8f vzero = {0.f, 0.f, 0.f, 0.f, 0.f, 0.f, 0.f, 0.f};

  for (int chunk = 0; chunk < (D_MODEL / 16) / (4 * NB); ++chunk) {
    const int ntBase = wc * 32 + chunk * NB;
    v8f C[NB];
    #pragma unroll
    for (int j = 0; j < NB; ++j) C[j] = vzero;

    for (int k0 = 0; k0 < D_MODEL; k0 += 32) {
      // ---- issue ALL loads for this k-step into distinct registers first ----
      // A tiles from LDS: ds_load_b128 x2 each (K k..k+7 and k+8..k+15)
      const v4u* ph = (const v4u*)&sAh[arow * D_MODEL + k0 + akb];
      const v4u* po = (const v4u*)&sAo[arow * D_MODEL + k0 + akb];
      const v4u ah0 = ph[0], ah1 = ph[1];
      const v4u ao0 = po[0], ao1 = po[1];

      // B tiles from global (L2-resident bf16 weights): contiguous 32 B / lane
      v4u bl0[NB], bl1[NB], ba0[NB], ba1[NB];
      #pragma unroll
      for (int j = 0; j < NB; ++j) {
        const int n = (ntBase + j) * 16 + lmod;
        const v4u* pl = (const v4u*)&wl[(size_t)n * D_MODEL + k0 + bkb];
        const v4u* pw = (const v4u*)&wa[(size_t)n * D_MODEL + k0 + bkb];
        bl0[j] = pl[0]; bl1[j] = pl[1];
        ba0[j] = pw[0]; ba1[j] = pw[1];
      }

      if (k0 + 512 < D_MODEL) {   // prefetch ~16 k-steps (1 KB) ahead
        __builtin_prefetch((const void*)&wl[(size_t)(ntBase * 16 + lmod) * D_MODEL + k0 + 512 + bkb], 0, 1);
        __builtin_prefetch((const void*)&wa[(size_t)(ntBase * 16 + lmod) * D_MODEL + k0 + 512 + bkb], 0, 1);
      }

      // ---- matrix ops: 8 wmma per k-step, loads above drain progressively ----
      const v16bf Ah = pack2(ah0, ah1);
      const v16bf Ao = pack2(ao0, ao1);
      #pragma unroll
      for (int j = 0; j < NB; ++j) {
        C[j] = wmma_bf16(Ah, pack2(bl0[j], bl1[j]), C[j]);
        C[j] = wmma_bf16(Ao, pack2(ba0[j], ba1[j]), C[j]);
      }
    }

    // ---- epilogue for this chunk: sigmoid, blend, LN stats, store fused ----
    #pragma unroll
    for (int j = 0; j < NB; ++j) {
      const int col = (ntBase + j) * 16 + lmod;
      const float bg  = bgate[col];
      const float g2c = g2v[col];
      const float b2c = b2v[col];
      const float at  = attnb[col];
      #pragma unroll
      for (int v = 0; v < 8; ++v) {
        const int rl = mt * 16 + lhalf * 8 + v;      // C layout: VGPR v -> row
        const size_t gidx = (size_t)(rowBase + rl) * D_MODEL + col;
        const float h  = h_llm[gidx];
        const float ao = (h + at - sMu[rl]) * sRs[rl] * g2c + b2c;
        const float logit = C[j][v] + bg;
        const float g = 1.f / (1.f + __expf(-logit));
        const float fused = g * ao + (1.f - g) * h;
        out[gidx] = fused;                            // unnormalized, pass 2 below
        s1acc[v] += fused;
        s2acc[v] += fused * fused;
      }
    }
  }

  // ---- final LN over the full row ----
  #pragma unroll
  for (int v = 0; v < 8; ++v) {
    const int rl = mt * 16 + lhalf * 8 + v;
    atomicAdd(&sS1[rl], s1acc[v]);
    atomicAdd(&sS2[rl], s2acc[v]);
  }
  __threadfence_block();
  __syncthreads();
  if (tid < MT) {
    const float mu = sS1[tid] * (1.f / D_MODEL);
    sMu3[tid] = mu;
    sRs3[tid] = rsqrtf(sS2[tid] * (1.f / D_MODEL) - mu * mu + 1e-5f);
  }
  __syncthreads();
  for (int idx = tid; idx < MT * D_MODEL; idx += NTHREADS) {
    const int rl = idx >> 11;
    const int c  = idx & (D_MODEL - 1);
    const size_t gidx = (size_t)(rowBase + rl) * D_MODEL + c;
    const float f = out[gidx];
    out[gidx] = (f - sMu3[rl]) * sRs3[rl] * g3v[c] + b3v[c];
  }
}

// ---------------------------------------------------------------------------
extern "C" void kernel_launch(void* const* d_in, const int* in_sizes, int n_in,
                              void* d_out, int out_size, void* d_ws, size_t ws_size,
                              hipStream_t stream) {
  (void)in_sizes; (void)n_in; (void)out_size; (void)ws_size;
  const float* h_llm    = (const float*)d_in[0];
  const float* h_change = (const float*)d_in[1];
  const float* w_proj   = (const float*)d_in[2];
  const float* b_proj   = (const float*)d_in[3];
  const float* g1       = (const float*)d_in[4];
  const float* b1       = (const float*)d_in[5];
  const float* w_v      = (const float*)d_in[6];
  const float* b_v      = (const float*)d_in[7];
  const float* w_o      = (const float*)d_in[8];
  const float* b_o      = (const float*)d_in[9];
  const float* g2       = (const float*)d_in[10];
  const float* b2       = (const float*)d_in[11];
  const float* w_gate   = (const float*)d_in[12];
  const float* b_gate   = (const float*)d_in[13];
  const float* g3       = (const float*)d_in[14];
  const float* b3       = (const float*)d_in[15];
  float* out = (float*)d_out;

  char* ws = (char*)d_ws;
  float*  hc   = (float*)(ws);                 // 4 x 2048
  float*  tve  = (float*)(ws + 32 * 1024);     // 4 x 2048
  float*  attn = (float*)(ws + 64 * 1024);     // 4 x 2048
  bf16_t* wl   = (bf16_t*)(ws + (1 << 20));                                  // 8 MB
  bf16_t* wa   = (bf16_t*)(ws + (1 << 20) + (size_t)D_MODEL * D_MODEL * 2);  // 8 MB

  proj_ln_kernel<<<4, NTHREADS, 0, stream>>>(h_change, w_proj, b_proj, g1, b1, hc);
  small_gemm_kernel<<<4, NTHREADS, 0, stream>>>(hc, w_v, b_v, tve);
  small_gemm_kernel<<<4, NTHREADS, 0, stream>>>(tve, w_o, b_o, attn);
  cvt_wgate_kernel<<<2048, NTHREADS, 0, stream>>>(w_gate, wl, wa);
  fused_gate_ln_kernel<<<(4 * 4096) / MT, NTHREADS, 0, stream>>>(
      h_llm, attn, wl, wa, g2, b2, b_gate, g3, b3, out);
}
